// StyleGenerator_79663053406282
// MI455X (gfx1250) — matrix operand order repaired and checked by reading, not compiled, not run on hardware
//
#include <hip/hip_runtime.h>
#include <cstdint>
#include <cstddef>

// ---------------------------------------------------------------------------
// CDNA5 (gfx1250) StyleGAN generator forward.
//  - conv lowered to implicit GEMM on V_WMMA_F32_16X16X32_F16 (wave32)
//  - per-batch modulated weights precomputed to f16 (K-major) so the conv
//    A-tile staging is a pure tile copy done with GLOBAL_LOAD_ASYNC_TO_LDS,
//    double-buffered in LDS, synchronized with s_wait_asynccnt.
//  - 4 co-subtiles per wave => 4 WMMAs per B-fragment build.
// ---------------------------------------------------------------------------

typedef __attribute__((ext_vector_type(16))) _Float16 v16h;
typedef __attribute__((ext_vector_type(8)))  _Float16 h8;
typedef __attribute__((ext_vector_type(8)))  float    v8f;

union F16Frag { v16h v; h8 g[2]; _Float16 h[16]; };
union F32Acc  { v8f  v; float f[8]; };

#define LRELU(x) ((x) > 0.0f ? (x) : 0.2f * (x))

// ----------------------------- noise (hash -> Box-Muller) -------------------
__device__ __forceinline__ unsigned hash_u32(unsigned x) {
    x ^= x >> 16; x *= 0x7feb352dU;
    x ^= x >> 15; x *= 0x846ca68bU;
    x ^= x >> 16;
    return x;
}

__device__ __forceinline__ float gauss_noise(int layer, int b, int pix) {
    unsigned s0 = hash_u32(0x9e3779b9u ^ (unsigned)(layer * 0x85ebca6b)
                                       ^ (unsigned)(b * 0xc2b2ae35) ^ (unsigned)pix);
    unsigned s1 = hash_u32(s0 + 0x68bc21ebu);
    float u1 = ((s0 >> 8) + 1u) * (1.0f / 16777217.0f);
    float u2 = (s1 >> 8) * (1.0f / 16777216.0f);
    return sqrtf(-2.0f * logf(u1)) * cosf(6.28318530718f * u2);
}

// ----------------------------- mapping network ------------------------------
__global__ void pixelnorm_kernel(const float* __restrict__ z, _Float16* __restrict__ w16) {
    __shared__ float red[512];
    int row = blockIdx.x;           // 16 blocks
    int t   = threadIdx.x;          // 512 threads
    float v = (row < 8) ? z[row * 512 + t] : 0.0f;
    red[t] = v * v;
    __syncthreads();
    for (int s = 256; s > 0; s >>= 1) {
        if (t < s) red[t] += red[t + s];
        __syncthreads();
    }
    float scale = rsqrtf(red[0] * (1.0f / 512.0f) + 1e-8f);
    w16[row * 512 + t] = (_Float16)(v * scale);
}

// wout = leaky(win @ Wm^T + b).  M=16 (8 valid), N=K=512. One wave / 16 cols.
__global__ __launch_bounds__(32) void mapping_gemm_kernel(
    const _Float16* __restrict__ win, const float* __restrict__ Wm,
    const float* __restrict__ bias, _Float16* __restrict__ wout,
    float* __restrict__ wfout)
{
    const int lane   = threadIdx.x & 31;
    const int n0     = blockIdx.x * 16;
    const int halfId = lane >> 4;
    const int mrow   = lane & 15;

    F16Frag a, bf;
    F32Acc  c;
#pragma unroll
    for (int r = 0; r < 8; r++) c.f[r] = 0.0f;

    for (int k0 = 0; k0 < 512; k0 += 32) {
#pragma unroll
        for (int j = 0; j < 16; j++) {
            int v  = j >> 1;
            int ka = ((v >= 4) ? 16 : 0) + halfId * 8 + (v & 3) * 2 + (j & 1);
            a.h[j] = win[mrow * 512 + k0 + ka];                     // A[m][k]
            int kb = ((j >= 8) ? 16 : 0) + halfId * 8 + (j & 7);
            bf.h[j] = (_Float16)Wm[(n0 + mrow) * 512 + k0 + kb];    // B[k][n] = Wm[n][k]
        }
        c.v = __builtin_amdgcn_wmma_f32_16x16x32_f16(false, a.v, false, bf.v,
                                                     (short)0, c.v, false, false);
    }
#pragma unroll
    for (int r = 0; r < 8; r++) {
        int m = r + halfId * 8;
        int n = n0 + (lane & 15);
        float val = c.f[r] + bias[n];
        val = LRELU(val);
        wout[m * 512 + n]  = (_Float16)val;
        wfout[m * 512 + n] = val;
    }
}

// ----------------------------- styles / demod -------------------------------
__global__ void style_kernel(const float* __restrict__ wlat, const float* __restrict__ mw,
                             const float* __restrict__ mb, float* __restrict__ sout, int Cin)
{
    int idx = blockIdx.x * blockDim.x + threadIdx.x;
    if (idx >= 8 * Cin) return;
    int b = idx / Cin, i = idx - b * Cin;
    float acc = 0.0f;
    const float* wr = wlat + b * 512;
    const float* mr = mw + (size_t)i * 512;
    for (int k = 0; k < 512; k++) acc += wr[k] * mr[k];
    sout[idx] = acc + mb[i] + 1.0f;
}

__global__ void w2_kernel(const float* __restrict__ W, float* __restrict__ w2,
                          int Cin, int Cout)
{
    int idx = blockIdx.x * blockDim.x + threadIdx.x;
    if (idx >= Cout * Cin) return;
    const float* p = W + (size_t)idx * 9;
    float acc = 0.0f;
#pragma unroll
    for (int t = 0; t < 9; t++) acc += p[t] * p[t];
    w2[idx] = acc;
}

__global__ void demod_kernel(const float* __restrict__ s, const float* __restrict__ w2,
                             float* __restrict__ dvec, int Cin, int Cout)
{
    int idx = blockIdx.x * blockDim.x + threadIdx.x;
    if (idx >= 8 * Cout) return;
    int b = idx / Cout, co = idx - b * Cout;
    const float* sr = s + b * Cin;
    const float* wr = w2 + (size_t)co * Cin;
    float acc = 0.0f;
    for (int ci = 0; ci < Cin; ci++) {
        float sv = sr[ci];
        acc += sv * sv * wr[ci];
    }
    dvec[idx] = rsqrtf(acc + 1e-8f);
}

// Wmod[b][co][k] = W[co][ci][t] * s[b][ci], k = ci*9+t, f16 K-major
__global__ void modweights_kernel(const float* __restrict__ W, const float* __restrict__ s,
                                  _Float16* __restrict__ Wmod, int Cin, int Cout)
{
    int per = Cout * Cin * 9;
    int idx = blockIdx.x * blockDim.x + threadIdx.x;
    if (idx >= 8 * per) return;
    int b = idx / per;
    int r = idx - b * per;
    int ci = (r / 9) % Cin;
    Wmod[idx] = (_Float16)(W[r] * s[b * Cin + ci]);
}

__global__ void const_init_kernel(const float* __restrict__ cst, _Float16* __restrict__ x) {
    int idx = blockIdx.x * blockDim.x + threadIdx.x;
    const int per = 512 * 16;
    if (idx >= 8 * per) return;
    x[idx] = (_Float16)cst[idx % per];
}

// ----------------------------- modulated 3x3 conv ---------------------------
// Block = 4 waves = 64co x 64pix macro-tile. Each wave: 16 pixels, 4 co-tiles.
// A tiles (64co x 32k f16, 4KB) async-copied global->LDS, double-buffered.
__global__ __launch_bounds__(128) void conv_mod_kernel(
    const _Float16* __restrict__ Wmod,   // [8][Cout][K]
    const _Float16* __restrict__ xin, _Float16* __restrict__ xout,
    const float* __restrict__ dvec, const float* __restrict__ nw,
    int Cin, int Cout, int Hout, int Wout, int up, int layer)
{
    __shared__ __align__(16) _Float16 As[2][64 * 32];

    const int tid    = threadIdx.x;
    const int lane   = tid & 31;
    const int wave   = tid >> 5;
    const int b      = blockIdx.z;
    const int co0    = blockIdx.y * 64;
    const int Npix   = Hout * Wout;
    const int halfId = lane >> 4;
    const int nIdx   = lane & 15;
    const int pix    = blockIdx.x * 64 + wave * 16 + nIdx;
    const bool pixOk = (pix < Npix);
    const int ph     = pixOk ? pix / Wout : 0;
    const int pw     = pixOk ? pix - ph * Wout : 0;
    const int Hin    = up ? (Hout >> 1) : Hout;
    const int Win    = up ? (Wout >> 1) : Wout;
    const _Float16* xinB  = xin + (size_t)b * Cin * Hin * Win;

    const int K = Cin * 9;                     // multiple of 32 for all layers
    const _Float16* WmodB = Wmod + ((size_t)b * Cout + co0) * (size_t)K;

    // async-stage one 64x32 A tile (4KB): 256 x 16B, 2 transfers per thread
    auto stage = [&](int k0, int buf) {
#pragma unroll
        for (int u = 0; u < 2; u++) {
            int id2 = tid * 2 + u;             // 0..255
            int row = id2 >> 2;                // 0..63 (co)
            int seg = id2 & 3;                 // 0..3 (8 f16 each)
            const _Float16* src = WmodB + (size_t)row * K + k0 + seg * 8;
            unsigned dst = (unsigned)(uintptr_t)&As[buf][row * 32 + seg * 8];
            asm volatile("global_load_async_to_lds_b128 %0, %1, off"
                         :: "v"(dst), "v"((unsigned long long)(uintptr_t)src)
                         : "memory");
        }
    };

    F32Acc acc[4];
#pragma unroll
    for (int t2 = 0; t2 < 4; t2++)
#pragma unroll
        for (int r = 0; r < 8; r++) acc[t2].f[r] = 0.0f;

    const int nCh = K / 32;
    stage(0, 0);
    for (int i = 0; i < nCh; i++) {
        const int k0 = i * 32;
        if (i + 1 < nCh) {
            stage(k0 + 32, (i + 1) & 1);
            asm volatile("s_wait_asynccnt 0x2" ::: "memory");   // chunk i landed
        } else {
            asm volatile("s_wait_asynccnt 0x0" ::: "memory");
        }
        __syncthreads();

        // ---- B fragment: x patches (ISA 7.12.2: n = lane%16, k half by lane/16)
        F16Frag bfr;
#pragma unroll
        for (int j = 0; j < 16; j++) {
            int kb = k0 + ((j >= 8) ? 16 : 0) + halfId * 8 + (j & 7);
            int ci = kb / 9;
            int t  = kb - ci * 9;
            int hh = ph + (t / 3) - 1;
            int ww = pw + (t % 3) - 1;
            _Float16 xv = (_Float16)0.0f;
            if (pixOk && hh >= 0 && hh < Hout && ww >= 0 && ww < Wout) {
                int hy = up ? (hh >> 1) : hh;
                int wx = up ? (ww >> 1) : ww;
                xv = xinB[((size_t)ci * Hin + hy) * Win + wx];
            }
            bfr.h[j] = xv;
        }

        // ---- 4 co-subtiles: A frag = two 16B LDS vector loads each ----
        const _Float16* asBuf = &As[i & 1][0];
#pragma unroll
        for (int t2 = 0; t2 < 4; t2++) {
            F16Frag a;
            const _Float16* base = asBuf + (t2 * 16 + nIdx) * 32 + halfId * 8;
            a.g[0] = *(const h8*)base;             // k = half*8 + 0..7
            a.g[1] = *(const h8*)(base + 16);      // k = 16 + half*8 + 0..7
            acc[t2].v = __builtin_amdgcn_wmma_f32_16x16x32_f16(
                false, a.v, false, bfr.v, (short)0, acc[t2].v, false, false);
        }
        __syncthreads();
    }

    // ---- epilogue: demod, noise, leaky, store f16 ----
    float nv = gauss_noise(layer, b, pix);
#pragma unroll
    for (int t2 = 0; t2 < 4; t2++) {
#pragma unroll
        for (int r = 0; r < 8; r++) {
            int m  = r + halfId * 8;
            int co = co0 + t2 * 16 + m;
            float y = acc[t2].f[r] * dvec[b * Cout + co] + nw[co] * nv;
            y = LRELU(y);
            if (pixOk)
                xout[((size_t)(b * Cout + co) * Hout + ph) * Wout + pw] = (_Float16)y;
        }
    }
}

// ----------------------------- ToRGB (1x1 modulated) ------------------------
__global__ void torgb_kernel(const _Float16* __restrict__ xin, const float* __restrict__ s,
                             const float* __restrict__ Wrgb, const float* __restrict__ rgbPrev,
                             float* __restrict__ rgbOut, int Cin, int H, int Wd, int doTanh)
{
    int idx = blockIdx.x * blockDim.x + threadIdx.x;
    int Np  = H * Wd;
    if (idx >= 8 * Np) return;
    int b = idx / Np, pix = idx - b * Np;
    int h = pix / Wd, w = pix - h * Wd;

    float a0 = 0.0f, a1 = 0.0f, a2 = 0.0f;
    const _Float16* xp = xin + (size_t)b * Cin * Np + pix;
    const float* sB = s + b * Cin;
    for (int i = 0; i < Cin; i++) {
        float f = (float)xp[(size_t)i * Np] * sB[i];
        a0 += f * Wrgb[i];
        a1 += f * Wrgb[Cin + i];
        a2 += f * Wrgb[2 * Cin + i];
    }
    float p0 = 0.0f, p1 = 0.0f, p2 = 0.0f;
    if (rgbPrev) {
        int Hh = H >> 1, Wh = Wd >> 1;
        size_t pb = (size_t)b * 3 * Hh * Wh + (size_t)(h >> 1) * Wh + (w >> 1);
        p0 = rgbPrev[pb];
        p1 = rgbPrev[pb + (size_t)Hh * Wh];
        p2 = rgbPrev[pb + 2 * (size_t)Hh * Wh];
    }
    float o0 = a0 + p0, o1 = a1 + p1, o2 = a2 + p2;
    if (doTanh) { o0 = tanhf(o0); o1 = tanhf(o1); o2 = tanhf(o2); }
    size_t ob = (size_t)b * 3 * Np + pix;
    rgbOut[ob] = o0;
    rgbOut[ob + Np] = o1;
    rgbOut[ob + 2 * (size_t)Np] = o2;
}

// ---------------------------------------------------------------------------
// Host orchestration
// ---------------------------------------------------------------------------
extern "C" void kernel_launch(void* const* d_in, const int* in_sizes, int n_in,
                              void* d_out, int out_size, void* d_ws, size_t ws_size,
                              hipStream_t stream)
{
    (void)in_sizes; (void)n_in; (void)out_size; (void)ws_size;

    const float* z    = (const float*)d_in[0];
    const float* mapw = (const float*)d_in[1];
    const float* mapb = (const float*)d_in[2];
    const float* cst  = (const float*)d_in[3];
    auto convW  = [&](int l) { return (const float*)d_in[4 + 4 * l + 0]; };
    auto convMW = [&](int l) { return (const float*)d_in[4 + 4 * l + 1]; };
    auto convMB = [&](int l) { return (const float*)d_in[4 + 4 * l + 2]; };
    auto convNW = [&](int l) { return (const float*)d_in[4 + 4 * l + 3]; };
    auto rgbW   = [&](int r) { return (const float*)d_in[56 + 3 * r + 0]; };
    auto rgbMW  = [&](int r) { return (const float*)d_in[56 + 3 * r + 1]; };
    auto rgbMB  = [&](int r) { return (const float*)d_in[56 + 3 * r + 2]; };

    static const int convCin[13]  = {512,512,512,512,512,512,512,512,256,256,128,128,64};
    static const int convCout[13] = {512,512,512,512,512,512,512,256,256,128,128,64,64};
    static const int rgbCinA[7]   = {512,512,512,512,256,128,64};

    // ---- workspace layout ----
    char* ws = (char*)d_ws;
    size_t off = 0;
    auto alloc = [&](size_t bytes) -> char* {
        char* p = ws + off;
        off = (off + bytes + 255) & ~(size_t)255;
        return p;
    };
    _Float16* w16a     = (_Float16*)alloc(16 * 512 * sizeof(_Float16));
    _Float16* w16b     = (_Float16*)alloc(16 * 512 * sizeof(_Float16));
    float*    wfin     = (float*)alloc(16 * 512 * sizeof(float));
    float*    styleBuf = (float*)alloc(8 * 7424 * sizeof(float));
    float*    w2buf    = (float*)alloc(512 * 512 * sizeof(float));
    float*    dbuf     = (float*)alloc(8 * 512 * sizeof(float));
    _Float16* wmodBuf  = (_Float16*)alloc((size_t)8 * 512 * 512 * 9 * sizeof(_Float16));
    const size_t XMAX  = (size_t)8 * 64 * 256 * 256;     // largest activation
    _Float16* xA       = (_Float16*)alloc(XMAX * sizeof(_Float16));
    _Float16* xB       = (_Float16*)alloc(XMAX * sizeof(_Float16));
    const size_t RGBN  = (size_t)8 * 3 * 256 * 256;
    float*    rgbA     = (float*)alloc(RGBN * sizeof(float));
    float*    rgbB     = (float*)alloc(RGBN * sizeof(float));

    int convSoff[13], rgbSoff[7];
    int cur = 0;
    for (int l = 0; l < 13; l++) { convSoff[l] = cur; cur += 8 * convCin[l]; }
    for (int r = 0; r < 7;  r++) { rgbSoff[r]  = cur; cur += 8 * rgbCinA[r]; }

    // ---- mapping network ----
    pixelnorm_kernel<<<16, 512, 0, stream>>>(z, w16a);
    _Float16* wi = w16a; _Float16* wo = w16b;
    for (int i = 0; i < 8; i++) {
        mapping_gemm_kernel<<<32, 32, 0, stream>>>(wi, mapw + (size_t)i * 512 * 512,
                                                   mapb + i * 512, wo, wfin);
        _Float16* t = wi; wi = wo; wo = t;
    }

    // ---- styles ----
    for (int l = 0; l < 13; l++) {
        int n = 8 * convCin[l];
        style_kernel<<<(n + 255) / 256, 256, 0, stream>>>(wfin, convMW(l), convMB(l),
                                                          styleBuf + convSoff[l], convCin[l]);
    }
    for (int r = 0; r < 7; r++) {
        int n = 8 * rgbCinA[r];
        style_kernel<<<(n + 255) / 256, 256, 0, stream>>>(wfin, rgbMW(r), rgbMB(r),
                                                          styleBuf + rgbSoff[r], rgbCinA[r]);
    }

    // ---- constant input ----
    {
        int n = 8 * 512 * 16;
        const_init_kernel<<<(n + 255) / 256, 256, 0, stream>>>(cst, xA);
    }
    _Float16* xcur = xA; _Float16* xnxt = xB;

    auto run_conv = [&](int l, int size, int up) {
        int Cin = convCin[l], Cout = convCout[l];
        int n2 = Cout * Cin;
        w2_kernel<<<(n2 + 255) / 256, 256, 0, stream>>>(convW(l), w2buf, Cin, Cout);
        int nd = 8 * Cout;
        demod_kernel<<<(nd + 255) / 256, 256, 0, stream>>>(styleBuf + convSoff[l], w2buf,
                                                           dbuf, Cin, Cout);
        int nm = 8 * Cout * Cin * 9;
        modweights_kernel<<<(nm + 255) / 256, 256, 0, stream>>>(convW(l),
                                                                styleBuf + convSoff[l],
                                                                wmodBuf, Cin, Cout);
        int Np = size * size;
        dim3 grid((Np + 63) / 64, Cout / 64, 8);
        conv_mod_kernel<<<grid, 128, 0, stream>>>(wmodBuf, xcur, xnxt, dbuf, convNW(l),
                                                  Cin, Cout, size, size, up, l);
        _Float16* t = xcur; xcur = xnxt; xnxt = t;
    };

    // ---- 4x4 block ----
    run_conv(0, 4, 0);
    {
        int n = 8 * 16;
        torgb_kernel<<<(n + 255) / 256, 256, 0, stream>>>(xcur, styleBuf + rgbSoff[0],
                                                          rgbW(0), nullptr, rgbA,
                                                          rgbCinA[0], 4, 4, 0);
    }
    float* rcur = rgbA; float* rnxt = rgbB;

    // ---- 8x8 .. 256x256 blocks ----
    int li = 1;
    for (int blk = 0; blk < 6; blk++) {
        int size = 8 << blk;
        run_conv(li, size, 1); li++;
        run_conv(li, size, 0); li++;
        int r = blk + 1;
        bool last = (size == 256);
        float* outp = last ? (float*)d_out : rnxt;
        int n = 8 * size * size;
        torgb_kernel<<<(n + 255) / 256, 256, 0, stream>>>(xcur, styleBuf + rgbSoff[r],
                                                          rgbW(r), rcur, outp,
                                                          rgbCinA[r], size, size,
                                                          last ? 1 : 0);
        if (!last) { float* t = rcur; rcur = rnxt; rnxt = t; }
    }
}
